// GraphAttentionLayer_81698867904899
// MI455X (gfx1250) — compile-verified, matrix operand order checked by reading, not run on hardware
//
#include <hip/hip_runtime.h>
#include <hip/hip_bf16.h>

// Fused GAT layer for MI455X (gfx1250, wave32, WMMA).
// B=8, N=2048, DIN=128, DOUT=64.

typedef __attribute__((ext_vector_type(16))) _Float16 v16h;
typedef __attribute__((ext_vector_type(8)))  _Float16 v8h;
typedef __attribute__((ext_vector_type(8)))  float    v8f;
typedef __attribute__((ext_vector_type(4)))  float    f4;
typedef __attribute__((ext_vector_type(4)))  int      i4;

#define GAT_B 8
#define GAT_N 2048
#define GAT_DIN 128
#define GAT_D 64
#define GAT_ALPHA 0.2f
#define GAT_NEGBIG (-1.0e4f)   // expf(-1e4) == 0.0f; select target for masked entries

__device__ __forceinline__ float lane_bcast_f(float v, int srclane) {
    return __int_as_float(__builtin_amdgcn_ds_bpermute(srclane << 2, __float_as_int(v)));
}

// ---------------------------------------------------------------------------
// Pass 1: z = h@W (f32 in regs), s1[n]=z.a1, s2'[n]=mask[n]?z.a2:-1e30,
//         zT[b][d][n] (f16, transposed for WMMA B operand),
//         mpart[b][blk][d] = 32-row partial column sums of z (for meanz).
// grid (B, N/32), block 256.
// ---------------------------------------------------------------------------
__global__ __launch_bounds__(256)
void gat_pass1(const float* __restrict__ h, const int* __restrict__ mask,
               const float* __restrict__ W, const float* __restrict__ a,
               _Float16* __restrict__ zT, float* __restrict__ s1,
               float* __restrict__ s2p, float* __restrict__ mpart)
{
    __shared__ float Ws[GAT_DIN * GAT_D];   // 32 KB
    __shared__ float hs[32 * GAT_DIN];      // 16 KB
    __shared__ float red1[32][GAT_D];       // 8 KB
    __shared__ float red2[32][GAT_D];       // 8 KB
    __shared__ float mz[4][GAT_D];          // 1 KB

    const int b   = blockIdx.x;
    const int n0  = blockIdx.y * 32;
    const int tid = threadIdx.x;

    for (int idx = tid; idx < GAT_DIN * GAT_D; idx += 256) Ws[idx] = W[idx];
    const float* hb = h + ((size_t)b * GAT_N + n0) * GAT_DIN;
    for (int idx = tid; idx < 32 * GAT_DIN; idx += 256) hs[idx] = hb[idx];
    __syncthreads();

    const int d = tid & 63;
    const int g = tid >> 6;                 // 0..3
    const float a1d = a[d];
    const float a2d = a[GAT_D + d];
    float msum = 0.0f;

    #pragma unroll 1
    for (int t = 0; t < 8; ++t) {
        const int r = g + 4 * t;
        float z = 0.0f;
        #pragma unroll 8
        for (int k = 0; k < GAT_DIN; ++k)
            z = fmaf(hs[r * GAT_DIN + k], Ws[k * GAT_D + d], z);
        red1[r][d] = z * a1d;
        red2[r][d] = z * a2d;
        msum += z;
        zT[((size_t)b * GAT_D + d) * GAT_N + (n0 + r)] = (_Float16)z;
    }
    mz[g][d] = msum;
    __syncthreads();

    if (tid < GAT_D) {
        float s = mz[0][tid] + mz[1][tid] + mz[2][tid] + mz[3][tid];
        mpart[((size_t)b * 64 + blockIdx.y) * GAT_D + tid] = s;
    }
    if (tid < 32) {
        float ss1 = 0.0f, ss2 = 0.0f;
        #pragma unroll 8
        for (int dd = 0; dd < GAT_D; ++dd) { ss1 += red1[tid][dd]; ss2 += red2[tid][dd]; }
        const int n = n0 + tid;
        s1[(size_t)b * GAT_N + n] = ss1;
        const int mk = mask[(size_t)b * GAT_N + n];
        s2p[(size_t)b * GAT_N + n] = (mk > 0) ? ss2 : -1.0e30f;
    }
}

// ---------------------------------------------------------------------------
// Reduce: s2max[b] = max_j s2'[b][j];  meanz[b][d] = (1/N) * sum_blk mpart.
// grid B, block 256.
// ---------------------------------------------------------------------------
__global__ __launch_bounds__(256)
void gat_reduce(const float* __restrict__ s2p, const float* __restrict__ mpart,
                float* __restrict__ meanz, float* __restrict__ s2max)
{
    __shared__ float red[256];
    const int b = blockIdx.x, tid = threadIdx.x;
    float mx = -3.0e38f;
    for (int j = tid; j < GAT_N; j += 256) mx = fmaxf(mx, s2p[(size_t)b * GAT_N + j]);
    red[tid] = mx;
    __syncthreads();
    if (tid == 0) {
        float m = red[0];
        for (int i = 1; i < 256; ++i) m = fmaxf(m, red[i]);
        s2max[b] = m;
    }
    if (tid < GAT_D) {
        float s = 0.0f;
        for (int blk = 0; blk < 64; ++blk)
            s += mpart[((size_t)b * 64 + blk) * GAT_D + tid];
        meanz[b * GAT_D + tid] = s * (1.0f / (float)GAT_N);
    }
}

// ---------------------------------------------------------------------------
// Pass 2: fused masked-softmax attention + PV via WMMA.
// One wave per 16-row i-tile; 8 waves/block. grid (B, N/16/8)=(8,16), block 256.
// Per j-step of 32 columns: build f16 A tile P[16x32] in exact WMMA layout,
// load 4 B tiles (32x16 f16) from zT, 4x v_wmma_f32_16x16x32_f16.
// Validity is folded into the exp argument -> branch-free inner loop.
// ---------------------------------------------------------------------------
__global__ __launch_bounds__(256)
void gat_pass2(const int* __restrict__ adj, const int* __restrict__ mask,
               const float* __restrict__ s1, const float* __restrict__ s2p,
               const float* __restrict__ s2max, const float* __restrict__ meanz,
               const _Float16* __restrict__ zT, float* __restrict__ out)
{
    const int b    = blockIdx.x;
    const int wave = threadIdx.x >> 5;
    const int lane = threadIdx.x & 31;
    const int hi   = lane >> 4;             // 0/1 half of wave
    const int il   = lane & 15;             // row-in-tile (A.M) / col (B.N, C.N)
    const int c    = hi << 3;               // K-subset base offset
    const int i0   = (blockIdx.y * 8 + wave) * 16;
    const int i    = i0 + il;

    const float s1v   = s1[(size_t)b * GAT_N + i];
    const int   maski = mask[(size_t)b * GAT_N + i];
    const float smax  = s2max[b];
    float Mi = s1v + smax;
    Mi = (Mi > 0.0f) ? Mi : GAT_ALPHA * Mi;
    const bool rowvalid = (maski > 0);

    const int*      adjrow = adj + ((size_t)b * GAT_N + i) * GAT_N;
    const float*    s2b    = s2p + (size_t)b * GAT_N;
    const _Float16* zTb    = zT + (size_t)b * GAT_D * GAT_N;

    const _Float16* zrow0 = zTb + (size_t)(0 * 16 + il) * GAT_N;
    const _Float16* zrow1 = zTb + (size_t)(1 * 16 + il) * GAT_N;
    const _Float16* zrow2 = zTb + (size_t)(2 * 16 + il) * GAT_N;
    const _Float16* zrow3 = zTb + (size_t)(3 * 16 + il) * GAT_N;

    v8f c0 = {}, c1 = {}, c2 = {}, c3 = {};
    float lsum0 = 0.0f, lsum1 = 0.0f;

    #pragma unroll 2
    for (int j0 = 0; j0 < GAT_N; j0 += 32) {
        const int* ap = adjrow + j0 + c;
        __builtin_prefetch((const void*)(adjrow + j0 + 512), 0, 1);
        // adj is 134 MB streamed exactly once: non-temporal so zT/s2 stay cached.
        i4 A0 = __builtin_nontemporal_load((const i4*)(ap));
        i4 A1 = __builtin_nontemporal_load((const i4*)(ap + 4));
        i4 A2 = __builtin_nontemporal_load((const i4*)(ap + 16));
        i4 A3 = __builtin_nontemporal_load((const i4*)(ap + 20));
        const float* sp = s2b + j0 + c;
        f4 S0 = *(const f4*)(sp);
        f4 S1 = *(const f4*)(sp + 4);
        f4 S2 = *(const f4*)(sp + 16);
        f4 S3 = *(const f4*)(sp + 20);

        const int av[16]   = {A0[0],A0[1],A0[2],A0[3], A1[0],A1[1],A1[2],A1[3],
                              A2[0],A2[1],A2[2],A2[3], A3[0],A3[1],A3[2],A3[3]};
        const float sv[16] = {S0[0],S0[1],S0[2],S0[3], S1[0],S1[1],S1[2],S1[3],
                              S2[0],S2[1],S2[2],S2[3], S3[0],S3[1],S3[2],S3[3]};

        v16h pa;
        #pragma unroll
        for (int q = 0; q < 16; ++q) {
            float e = s1v + sv[q];
            e = (e > 0.0f) ? e : GAT_ALPHA * e;
            // Branch-free: invalid entries get a huge-negative exponent -> p == 0.
            const bool valid = (av[q] > 0) && rowvalid;
            float arg = valid ? (e - Mi) : GAT_NEGBIG;
            float p = __expf(arg);
            _Float16 ph = (_Float16)p;
            pa[q] = ph;
            if (q & 1) lsum1 += (float)ph; else lsum0 += (float)ph;
        }

        // B tiles: zT rows are contiguous in j -> two 16B loads per d-tile.
        v8h lo0 = *(const v8h*)(zrow0 + j0 + c), hi0 = *(const v8h*)(zrow0 + j0 + c + 16);
        v8h lo1 = *(const v8h*)(zrow1 + j0 + c), hi1 = *(const v8h*)(zrow1 + j0 + c + 16);
        v8h lo2 = *(const v8h*)(zrow2 + j0 + c), hi2 = *(const v8h*)(zrow2 + j0 + c + 16);
        v8h lo3 = *(const v8h*)(zrow3 + j0 + c), hi3 = *(const v8h*)(zrow3 + j0 + c + 16);
        v16h B0 = __builtin_shufflevector(lo0, hi0, 0,1,2,3,4,5,6,7,8,9,10,11,12,13,14,15);
        v16h B1 = __builtin_shufflevector(lo1, hi1, 0,1,2,3,4,5,6,7,8,9,10,11,12,13,14,15);
        v16h B2 = __builtin_shufflevector(lo2, hi2, 0,1,2,3,4,5,6,7,8,9,10,11,12,13,14,15);
        v16h B3 = __builtin_shufflevector(lo3, hi3, 0,1,2,3,4,5,6,7,8,9,10,11,12,13,14,15);

        c0 = __builtin_amdgcn_wmma_f32_16x16x32_f16(false, pa, false, B0, (short)0, c0, false, false);
        c1 = __builtin_amdgcn_wmma_f32_16x16x32_f16(false, pa, false, B1, (short)0, c1, false, false);
        c2 = __builtin_amdgcn_wmma_f32_16x16x32_f16(false, pa, false, B2, (short)0, c2, false, false);
        c3 = __builtin_amdgcn_wmma_f32_16x16x32_f16(false, pa, false, B3, (short)0, c3, false, false);
    }

    // Row sums: lane λ holds partial for row il over its K-subset; combine λ and λ^16.
    const float lsum = lsum0 + lsum1;
    const float rowl = lsum + lane_bcast_f(lsum, lane ^ 16);

    float* outb = out + ((size_t)b * GAT_N + i0) * GAT_D;
    const float* mzb = meanz + b * GAT_D;

    #pragma unroll
    for (int r = 0; r < 8; ++r) {
        const int m = r + 8 * hi;           // output row within tile; m in [0,16)
        const float lm = lane_bcast_f(rowl, m); // row m's sum lives in lane m
        const bool uni = !(lm > 0.0f);      // fully-masked row -> uniform softmax
        const float inv = uni ? 0.0f : (1.0f / lm);
        #pragma unroll
        for (int t = 0; t < 4; ++t) {
            float v;
            switch (t) {
                case 0: v = c0[r]; break;
                case 1: v = c1[r]; break;
                case 2: v = c2[r]; break;
                default: v = c3[r]; break;
            }
            v *= inv;
            if (uni) v = mzb[t * 16 + il];
            v = (v > 0.0f) ? v : (__expf(v) - 1.0f);   // ELU
            outb[(size_t)m * GAT_D + t * 16 + il] = v;
        }
    }
}

// ---------------------------------------------------------------------------
extern "C" void kernel_launch(void* const* d_in, const int* in_sizes, int n_in,
                              void* d_out, int out_size, void* d_ws, size_t ws_size,
                              hipStream_t stream) {
    const float* h    = (const float*)d_in[0];
    const int*   adj  = (const int*)d_in[1];
    const int*   mask = (const int*)d_in[2];
    const float* W    = (const float*)d_in[3];
    const float* a    = (const float*)d_in[4];
    float* out = (float*)d_out;

    char* ws = (char*)d_ws;
    _Float16* zT  = (_Float16*)ws;                       // 2 MB
    float* s1     = (float*)(ws + (2u << 20));           // 64 KB
    float* s2p    = s1 + GAT_B * GAT_N;                  // 64 KB
    float* mpart  = s2p + GAT_B * GAT_N;                 // 128 KB
    float* meanz  = mpart + GAT_B * 64 * GAT_D;          // 2 KB
    float* s2max  = meanz + GAT_B * GAT_D;               // 32 B

    gat_pass1<<<dim3(GAT_B, GAT_N / 32), 256, 0, stream>>>(h, mask, W, a, zT, s1, s2p, mpart);
    gat_reduce<<<GAT_B, 256, 0, stream>>>(s2p, mpart, meanz, s2max);
    gat_pass2<<<dim3(GAT_B, GAT_N / (16 * 8)), 256, 0, stream>>>(
        adj, mask, s1, s2p, s2max, meanz, zT, out);
}
